// MultiCodebookVectorQuantizer_64742337020151
// MI455X (gfx1250) — matrix operand order repaired and checked by reading, not compiled, not run on hardware
//
#include <hip/hip_runtime.h>
#include <hip/hip_bf16.h>

// Problem constants
#define MROWS 65536      // 4096*16 flattened rows
#define DDIM  256        // embedding dim (K)
#define NE    1024       // codebook size (N)
#define BETA  0.25f

typedef __attribute__((ext_vector_type(16))) __bf16 v16bf;
typedef __attribute__((ext_vector_type(8)))  __bf16 v8bf;
typedef __attribute__((ext_vector_type(8)))  float  v8f;
typedef __attribute__((ext_vector_type(4)))  int    v4i;

#define GLOBAL_AS __attribute__((address_space(1)))
#define LDS_AS    __attribute__((address_space(3)))

// gfx1250 async global->LDS DMA path (ASYNCcnt-tracked), with sync fallback.
#if __has_builtin(__builtin_amdgcn_global_load_async_to_lds_b128) && \
    __has_builtin(__builtin_amdgcn_s_wait_asynccnt)
#define USE_ASYNC_LDS 1
#else
#define USE_ASYNC_LDS 0
#endif

// Workspace byte offsets
#define WS_EMB_BF16 0u          // 1024*256*2 = 524288 B
#define WS_ENORM    524288u     // 1024*4
#define WS_IDX      528384u     // 65536*4
#define WS_HIST     790528u     // 1024*4
#define WS_LOSS     794624u     // 4

// Output float offsets: [loss][z_q (16777216)][perplexity][indices (65536)]
#define OUT_ZQ   1
#define OUT_PERP 16777217
#define OUT_IDX  16777218

// -------------------------------------------------------------------------
// Kernel 1: emb fp32 -> bf16 (row major) + ||e||^2 ; zero hist/loss.
// -------------------------------------------------------------------------
__global__ __launch_bounds__(256) void vq_prep_kernel(
    const float* __restrict__ emb, __bf16* __restrict__ embb,
    float* __restrict__ enorm, unsigned int* __restrict__ hist,
    float* __restrict__ loss_acc) {
  const int tid  = threadIdx.x;
  const int lane = tid & 31;
  const int row  = blockIdx.x * 8 + (tid >> 5);   // 0..1023

  const float4* zr = (const float4*)emb + (size_t)row * 64;
  float4 f0 = zr[lane * 2 + 0];
  float4 f1 = zr[lane * 2 + 1];

  v8bf o;
  o[0] = (__bf16)f0.x; o[1] = (__bf16)f0.y; o[2] = (__bf16)f0.z; o[3] = (__bf16)f0.w;
  o[4] = (__bf16)f1.x; o[5] = (__bf16)f1.y; o[6] = (__bf16)f1.z; o[7] = (__bf16)f1.w;
  ((v8bf*)embb)[(size_t)row * 32 + lane] = o;

  float s = f0.x*f0.x + f0.y*f0.y + f0.z*f0.z + f0.w*f0.w
          + f1.x*f1.x + f1.y*f1.y + f1.z*f1.z + f1.w*f1.w;
  #pragma unroll
  for (int off = 16; off; off >>= 1) s += __shfl_xor(s, off, 32);
  if (lane == 0) enorm[row] = s;

  if (blockIdx.x == 0) {
    for (int i = tid; i < NE; i += 256) hist[i] = 0u;
    if (tid == 0) *loss_acc = 0.0f;
  }
}

// -------------------------------------------------------------------------
// Async (or sync-fallback) stage of one 64-code chunk into an LDS buffer.
// 256 threads x 8 transfers of 16 B; padded row stride 33 uint4 (528 B).
// -------------------------------------------------------------------------
__device__ __forceinline__ void stage_chunk(const uint4* __restrict__ gB,
                                            uint4* lds, int tid) {
  #pragma unroll
  for (int i = 0; i < 8; ++i) {
    const int li   = i * 256 + tid;
    const int code = li >> 5;
    const int c16  = li & 31;
#if USE_ASYNC_LDS
    __builtin_amdgcn_global_load_async_to_lds_b128(
        (GLOBAL_AS v4i*)(gB + code * 32 + c16),
        (LDS_AS v4i*)(lds + code * 33 + c16),
        /*offset=*/0, /*cpol=*/0);
#else
    lds[code * 33 + c16] = gB[code * 32 + c16];
#endif
  }
}

// -------------------------------------------------------------------------
// Kernel 2: distances + argmin via bf16 WMMA.
//  - 512 blocks x 8 waves; wave w owns rows [blk*128 + w*16, +16)
//  - A fragments (16x256 bf16) fully register-resident
//  - codebook double-buffered through LDS via async global->LDS DMA
//  - B fragments software-pipelined one K-step ahead of the WMMA quad
// d[m][n] = ||e_n||^2 - 2*(z_m . e_n)   (||z||^2 is argmin-invariant)
// -------------------------------------------------------------------------
__global__ __launch_bounds__(256) void vq_argmin_kernel(
    const float* __restrict__ z, const __bf16* __restrict__ embb,
    const float* __restrict__ enorm, int* __restrict__ outIdx) {
  __shared__ uint4 ldsB[2][64 * 33];              // 2 x 33792 B

  const int tid    = threadIdx.x;
  const int w      = tid >> 5;
  const int L      = tid & 31;
  const int halfHi = L >> 4;                      // 0: lanes 0-15, 1: lanes 16-31
  const int c      = L & 15;
  const int rowbase = blockIdx.x * 128 + w * 16;

  // ---- Load A fragments (ISA 16-bit A layout: interleaved 8-element K runs)
  v16bf a[8];
  {
    const float* zr = z + (size_t)(rowbase + c) * DDIM;
    const int off = halfHi ? 8 : 0;
    #pragma unroll
    for (int s = 0; s < 8; ++s) {
      const int k0 = s * 32 + off;
      float4 f0 = *(const float4*)(zr + k0);
      float4 f1 = *(const float4*)(zr + k0 + 4);
      float4 f2 = *(const float4*)(zr + k0 + 16);
      float4 f3 = *(const float4*)(zr + k0 + 20);
      v16bf t;
      t[0]=(__bf16)f0.x;  t[1]=(__bf16)f0.y;  t[2]=(__bf16)f0.z;  t[3]=(__bf16)f0.w;
      t[4]=(__bf16)f1.x;  t[5]=(__bf16)f1.y;  t[6]=(__bf16)f1.z;  t[7]=(__bf16)f1.w;
      t[8]=(__bf16)f2.x;  t[9]=(__bf16)f2.y;  t[10]=(__bf16)f2.z; t[11]=(__bf16)f2.w;
      t[12]=(__bf16)f3.x; t[13]=(__bf16)f3.y; t[14]=(__bf16)f3.z; t[15]=(__bf16)f3.w;
      a[s] = t;
    }
  }

  float minv[8];
  int   mini[8];
  #pragma unroll
  for (int j = 0; j < 8; ++j) { minv[j] = 3.4e38f; mini[j] = 0; }

  const uint4* gB = (const uint4*)embb;           // 32 uint4 per code row

  // prologue: stage chunk 0
  stage_chunk(gB, ldsB[0], tid);
#if USE_ASYNC_LDS
  __builtin_amdgcn_s_wait_asynccnt(0);
#endif
  __syncthreads();

  union BB { uint4 u[2]; v16bf v; };
  const int bcol = c * 33 + halfHi;               // lane's column base (uint4)

  for (int nc = 0; nc < 16; ++nc) {               // 16 chunks x 64 codes
    const uint4* cur = ldsB[nc & 1];
    if (nc < 15)                                  // overlap next stage w/ compute
      stage_chunk(gB + (size_t)(nc + 1) * 64 * 32, ldsB[(nc + 1) & 1], tid);

    float en[4];
    #pragma unroll
    for (int nt = 0; nt < 4; ++nt) en[nt] = enorm[nc * 64 + nt * 16 + c];

    v8f acc[4] = {};
    BB bcur[4], bnxt[4];
    #pragma unroll
    for (int nt = 0; nt < 4; ++nt) {              // preload K-step 0
      const int bb = bcol + nt * 16 * 33;
      bcur[nt].u[0] = cur[bb];
      bcur[nt].u[1] = cur[bb + 2];
    }
    #pragma unroll
    for (int s = 0; s < 8; ++s) {                 // K = 256 in steps of 32
      if (s < 7) {                                // prefetch K-step s+1
        #pragma unroll
        for (int nt = 0; nt < 4; ++nt) {
          const int bb = bcol + nt * 16 * 33 + (s + 1) * 4;
          bnxt[nt].u[0] = cur[bb];
          bnxt[nt].u[1] = cur[bb + 2];
        }
      }
      #pragma unroll
      for (int nt = 0; nt < 4; ++nt)
        acc[nt] = __builtin_amdgcn_wmma_f32_16x16x32_bf16(
            false, a[s], false, bcur[nt].v, (short)0, acc[nt], false, false);
      #pragma unroll
      for (int nt = 0; nt < 4; ++nt) bcur[nt] = bnxt[nt];
    }

    #pragma unroll
    for (int nt = 0; nt < 4; ++nt) {
      const int n = nc * 64 + nt * 16 + c;
      #pragma unroll
      for (int j = 0; j < 8; ++j) {
        const float d = en[nt] - 2.0f * acc[nt][j];
        if (d < minv[j]) { minv[j] = d; mini[j] = n; }
      }
    }

#if USE_ASYNC_LDS
    if (nc < 15) __builtin_amdgcn_s_wait_asynccnt(0);
#endif
    __syncthreads();
  }

  // reduce argmin across the 16-lane N-group; ties -> lowest index (jnp.argmin)
  #pragma unroll
  for (int off = 8; off; off >>= 1) {
    #pragma unroll
    for (int j = 0; j < 8; ++j) {
      const float ov = __shfl_xor(minv[j], off, 16);
      const int   oi = __shfl_xor(mini[j], off, 16);
      if (ov < minv[j] || (ov == minv[j] && oi < mini[j])) { minv[j] = ov; mini[j] = oi; }
    }
  }
  if (c == 0) {
    #pragma unroll
    for (int j = 0; j < 8; ++j)
      outIdx[rowbase + j + 8 * halfHi] = mini[j];
  }
}

// -------------------------------------------------------------------------
// Kernel 3: gather z_q = emb[idx] (fp32), write z_q + index, accumulate
// sum((z - z_q)^2) and histogram. One block (256 thr) per row.
// -------------------------------------------------------------------------
__global__ __launch_bounds__(256) void vq_gather_kernel(
    const float* __restrict__ z, const float* __restrict__ emb,
    const int* __restrict__ idx, unsigned int* __restrict__ hist,
    float* __restrict__ loss_acc, float* __restrict__ out) {
  __shared__ float red[8];
  const int row  = blockIdx.x;
  const int t    = threadIdx.x;
  const int lane = t & 31;
  const int wv   = t >> 5;

  const int  n  = idx[row];
  const float zq = emb[(size_t)n * DDIM + t];
  const float zv = z[(size_t)row * DDIM + t];
  out[OUT_ZQ + (size_t)row * DDIM + t] = zq;

  float d = (zv - zq) * (zv - zq);
  #pragma unroll
  for (int off = 16; off; off >>= 1) d += __shfl_xor(d, off, 32);
  if (lane == 0) red[wv] = d;
  __syncthreads();
  if (t == 0) {
    float s = 0.f;
    #pragma unroll
    for (int i = 0; i < 8; ++i) s += red[i];
    atomicAdd(loss_acc, s);
    atomicAdd(&hist[n], 1u);
    out[OUT_IDX + row] = (float)n;
  }
}

// -------------------------------------------------------------------------
// Kernel 4: perplexity + final loss. 1 block x 1024 threads.
// -------------------------------------------------------------------------
__global__ __launch_bounds__(1024) void vq_final_kernel(
    const unsigned int* __restrict__ hist, const float* __restrict__ loss_acc,
    float* __restrict__ out) {
  __shared__ float red[32];
  const int t    = threadIdx.x;
  const int lane = t & 31;
  const int wv   = t >> 5;

  const float p = (float)hist[t] * (1.0f / (float)MROWS);
  float term = p * logf(p + 1e-10f);
  #pragma unroll
  for (int off = 16; off; off >>= 1) term += __shfl_xor(term, off, 32);
  if (lane == 0) red[wv] = term;
  __syncthreads();
  if (t < 32) {
    float s = red[t];
    #pragma unroll
    for (int off = 16; off; off >>= 1) s += __shfl_xor(s, off, 32);
    if (t == 0) {
      out[OUT_PERP] = expf(-s);
      out[0] = (1.0f + BETA) * (*loss_acc) * (1.0f / (float)(MROWS * DDIM));
    }
  }
}

// -------------------------------------------------------------------------
extern "C" void kernel_launch(void* const* d_in, const int* in_sizes, int n_in,
                              void* d_out, int out_size, void* d_ws, size_t ws_size,
                              hipStream_t stream) {
  const float* z   = (const float*)d_in[0];   // (4096,16,256) f32
  const float* emb = (const float*)d_in[1];   // (1024,256) f32
  float* out = (float*)d_out;

  char* ws = (char*)d_ws;
  __bf16*       embb     = (__bf16*)(ws + WS_EMB_BF16);
  float*        enorm    = (float*)(ws + WS_ENORM);
  int*          idx      = (int*)(ws + WS_IDX);
  unsigned int* hist     = (unsigned int*)(ws + WS_HIST);
  float*        loss_acc = (float*)(ws + WS_LOSS);

  vq_prep_kernel<<<128, 256, 0, stream>>>(emb, embb, enorm, hist, loss_acc);
  vq_argmin_kernel<<<MROWS / 128, 256, 0, stream>>>(z, embb, enorm, idx);
  vq_gather_kernel<<<MROWS, 256, 0, stream>>>(z, emb, idx, hist, loss_acc, out);
  vq_final_kernel<<<1, 1024, 0, stream>>>(hist, loss_acc, out);
}
